// GPT_3161095930008
// MI455X (gfx1250) — matrix operand order repaired and checked by reading, not compiled, not run on hardware
//
#include <hip/hip_runtime.h>
#include <hip/hip_bf16.h>
#include <math.h>

// GPT dims (fixed by reference)
#define BGPT 4
#define TGPT 1024
#define CGPT 512
#define HGPT 8
#define DGPT 64
#define LGPT 4
#define VGPT 16384
#define FFGPT 2048
#define MGPT (BGPT * TGPT)   // 4096 token rows

typedef __bf16 bf16_t;
typedef __attribute__((ext_vector_type(16))) __bf16 v16bf;
typedef __attribute__((ext_vector_type(8)))  float  v8f;

// ---------- helpers ----------

__device__ __forceinline__ unsigned short f2bf(float f) {
    union { float f; unsigned u; } v; v.f = f;
    unsigned u = v.u;
    unsigned r = u + 0x7FFFu + ((u >> 16) & 1u);   // round-to-nearest-even
    return (unsigned short)(r >> 16);
}

__device__ __forceinline__ v8f wmma_bf16(v16bf a, v16bf b, v8f c) {
    return __builtin_amdgcn_wmma_f32_16x16x32_bf16(
        /*neg_a=*/false, a, /*neg_b=*/false, b,
        /*c_mod=*/(short)0, c, /*reuse_a=*/false, /*reuse_b=*/false);
}

// A-fragment (16x32 bf16, row-major source, leading dim ld):
// lane: M = lane&15, kb = (lane>>4)*8; elems 0..7 -> K=kb+e, 8..15 -> K=kb+16+e
__device__ __forceinline__ v16bf load_a_global(const unsigned short* base, int ld) {
    int lane = threadIdx.x & 31;
    const bf16_t* p = reinterpret_cast<const bf16_t*>(base)
                      + (size_t)(lane & 15) * ld + ((lane >> 4) << 3);
    v16bf r;
#pragma unroll
    for (int e = 0; e < 8; ++e) { r[e] = p[e]; r[8 + e] = p[16 + e]; }
    return r;
}

// B-fragment (32x16) from LDS tile bt[row][BLD]: N = lane&15, kb = (lane>>4)*8; elems 0..7 -> K=kb+e, 8..15 -> K=kb+16+e
__device__ __forceinline__ v16bf load_b_lds(const unsigned short* bt, int row0,
                                            int n0, int bld) {
    int lane = threadIdx.x & 31;
    const bf16_t* p = reinterpret_cast<const bf16_t*>(bt)
                      + (size_t)(row0 + ((lane >> 4) << 3)) * bld + n0 + (lane & 15);
    v16bf r;
#pragma unroll
    for (int e = 0; e < 8; ++e) { r[e] = p[e * bld]; r[8 + e] = p[(16 + e) * bld]; }
    return r;
}

// ---------- kernels ----------

__global__ void k_embed(const int* __restrict__ idx, const float* __restrict__ tok,
                        const float* __restrict__ pos, float* __restrict__ x, long n) {
    long i = (long)blockIdx.x * blockDim.x + threadIdx.x;
    if (i >= n) return;
    long c  = i % CGPT;
    long bt = i / CGPT;
    long t  = bt % TGPT;
    x[i] = tok[(size_t)idx[bt] * CGPT + c] + pos[t * CGPT + c];
}

// Convert fp32 [N,K] -> bf16 transposed [K,N], tiled 64x64 through LDS.
// grid: (K/64, N/64, layers); both global phases fully coalesced.
__global__ __launch_bounds__(256) void k_convT(const float* __restrict__ W,
                                               unsigned short* __restrict__ Wt,
                                               int N, int K) {
    __shared__ unsigned short lt[64 * 72];
    size_t lay = (size_t)blockIdx.z * N * K;
    int k0 = blockIdx.x * 64;
    int n0 = blockIdx.y * 64;
#pragma unroll
    for (int e = 0; e < 16; ++e) {
        int idx = threadIdx.x + e * 256;
        int r = idx >> 6, c = idx & 63;            // r: n-local, c: k-local
        lt[r * 72 + c] = f2bf(W[lay + (size_t)(n0 + r) * K + k0 + c]);
    }
    __syncthreads();
#pragma unroll
    for (int e = 0; e < 16; ++e) {
        int idx = threadIdx.x + e * 256;
        int r = idx >> 6, c = idx & 63;            // r: k-local, c: n-local
        Wt[lay + (size_t)(k0 + r) * N + n0 + c] = lt[c * 72 + r];
    }
}

// Per-head K transpose: [B,T,C] bf16 -> [B,H,D,T] bf16, tiled 64x64.
__global__ __launch_bounds__(256) void k_transpose_k(const unsigned short* __restrict__ Kin,
                                                     unsigned short* __restrict__ Kout) {
    __shared__ unsigned short lt[64 * 72];
    int t0 = blockIdx.x * 64;
    int h  = blockIdx.y;
    int b  = blockIdx.z;
    size_t inBase  = ((size_t)b * TGPT + t0) * CGPT + (size_t)h * DGPT;
    size_t outBase = (((size_t)b * HGPT + h) * DGPT) * TGPT + t0;
#pragma unroll
    for (int e = 0; e < 16; ++e) {
        int idx = threadIdx.x + e * 256;
        int r = idx >> 6, c = idx & 63;            // r: t-local, c: d
        lt[r * 72 + c] = Kin[inBase + (size_t)r * CGPT + c];
    }
    __syncthreads();
#pragma unroll
    for (int e = 0; e < 16; ++e) {
        int idx = threadIdx.x + e * 256;
        int r = idx >> 6, c = idx & 63;            // r: d, c: t-local
        Kout[outBase + (size_t)r * TGPT + c] = lt[c * 72 + r];
    }
}

// One wave per row of C=512: 16 elems/lane, wave32 shuffle reduction.
__global__ __launch_bounds__(32) void k_layernorm(const float* __restrict__ x,
                                                  const float* __restrict__ w,
                                                  const float* __restrict__ b,
                                                  unsigned short* __restrict__ out) {
    int row  = blockIdx.x;
    int lane = threadIdx.x;
    const float* xr = x + (size_t)row * CGPT;
    float vals[16];
    float s = 0.f, s2 = 0.f;
#pragma unroll
    for (int i = 0; i < 16; ++i) {
        float v = xr[lane + i * 32];
        vals[i] = v; s += v; s2 += v * v;
    }
#pragma unroll
    for (int o = 16; o >= 1; o >>= 1) {
        s  += __shfl_xor(s,  o, 32);
        s2 += __shfl_xor(s2, o, 32);
    }
    float mu  = s * (1.0f / CGPT);
    float var = s2 * (1.0f / CGPT) - mu * mu;
    float rst = rsqrtf(var + 1e-5f);
    unsigned short* orow = out + (size_t)row * CGPT;
#pragma unroll
    for (int i = 0; i < 16; ++i) {
        int c = lane + i * 32;
        orow[c] = f2bf((vals[i] - mu) * rst * w[c] + b[c]);
    }
}

// GEMM: out[M,N] = act(A[M,K](bf16) @ Wt[K,N](bf16) + bias) (+resid fp32)
// Block: 256 threads = 8 waves (4 M-waves x 2 N-waves); tile 128x64, K step 64.
// Double-buffered LDS weight tiles: next-tile global loads issue before the
// WMMAs of the current tile, ds_stores go to the alternate buffer, and a
// single barrier per K-step separates producers from next-step consumers.
#define GTM 128
#define GTN 64
#define BLD 72   // LDS row pitch: 144B, 16B aligned, bank-spread

__global__ __launch_bounds__(256) void k_gemm(
    const unsigned short* __restrict__ A, const unsigned short* __restrict__ Wt,
    const float* __restrict__ bias, const float* __restrict__ resid,
    float* __restrict__ outF, unsigned short* __restrict__ outH,
    int M, int N, int K, int doGelu) {
    __shared__ unsigned short bt[2][64 * BLD];
    int tid  = threadIdx.x;
    int wave = tid >> 5, lane = tid & 31;
    int wm = wave & 3, wn = wave >> 2;
    int rowBlk = blockIdx.x * GTM, colBlk = blockIdx.y * GTN;
    int skk = tid >> 3;            // 0..31 (k row within slab)
    int sn  = (tid & 7) << 3;      // 0..56 (8-elem n chunk)
    const unsigned short* wsrc = Wt + (size_t)skk * N + colBlk + sn;

    v8f acc[2][2];
#pragma unroll
    for (int mi = 0; mi < 2; ++mi)
#pragma unroll
        for (int ni = 0; ni < 2; ++ni)
#pragma unroll
            for (int i = 0; i < 8; ++i) acc[mi][ni][i] = 0.f;

    // prologue: stage first 64x64 weight tile into buffer 0
    {
        uint4 w0 = *reinterpret_cast<const uint4*>(wsrc);
        uint4 w1 = *reinterpret_cast<const uint4*>(wsrc + (size_t)32 * N);
        *reinterpret_cast<uint4*>(&bt[0][skk * BLD + sn]) = w0;
        *reinterpret_cast<uint4*>(&bt[0][(skk + 32) * BLD + sn]) = w1;
    }
    __syncthreads();

    int buf = 0;
    for (int k0 = 0; k0 < K; k0 += 64) {
        int hasNext = (k0 + 64 < K);
        uint4 w0, w1;
        if (hasNext) {   // issue next-tile loads early; latency hidden by WMMAs
            w0 = *reinterpret_cast<const uint4*>(wsrc + (size_t)(k0 + 64) * N);
            w1 = *reinterpret_cast<const uint4*>(wsrc + (size_t)(k0 + 96) * N);
            if (k0 + 128 < K)
                __builtin_prefetch(wsrc + (size_t)(k0 + 128) * N, 0, 2);
        }

        const unsigned short* btc = &bt[buf][0];
        v16bf af[2][2], bfr[2][2];
#pragma unroll
        for (int mi = 0; mi < 2; ++mi)
#pragma unroll
            for (int kc = 0; kc < 2; ++kc)
                af[mi][kc] = load_a_global(
                    A + (size_t)(rowBlk + wm * 32 + mi * 16) * K + k0 + kc * 32, K);
#pragma unroll
        for (int ni = 0; ni < 2; ++ni)
#pragma unroll
            for (int kc = 0; kc < 2; ++kc)
                bfr[ni][kc] = load_b_lds(btc, kc * 32, wn * 32 + ni * 16, BLD);
#pragma unroll
        for (int kc = 0; kc < 2; ++kc)
#pragma unroll
            for (int mi = 0; mi < 2; ++mi)
#pragma unroll
                for (int ni = 0; ni < 2; ++ni)
                    acc[mi][ni] = wmma_bf16(af[mi][kc], bfr[ni][kc], acc[mi][ni]);

        if (hasNext) {   // store next tile into the alternate buffer
            *reinterpret_cast<uint4*>(&bt[buf ^ 1][skk * BLD + sn]) = w0;
            *reinterpret_cast<uint4*>(&bt[buf ^ 1][(skk + 32) * BLD + sn]) = w1;
        }
        __syncthreads();
        buf ^= 1;
    }

    // epilogue: C-frag element (r,lane) -> M = r + (lane>=16)*8, N = lane&15
    int half = lane >> 4;
    int nl   = lane & 15;
#pragma unroll
    for (int mi = 0; mi < 2; ++mi)
#pragma unroll
        for (int ni = 0; ni < 2; ++ni) {
            int nc = colBlk + wn * 32 + ni * 16 + nl;
#pragma unroll
            for (int r = 0; r < 8; ++r) {
                int mr = rowBlk + wm * 32 + mi * 16 + r + half * 8;
                float v = acc[mi][ni][r];
                if (bias)   v += bias[nc];
                if (doGelu) v = 0.5f * v * (1.0f + erff(v * 0.70710678118f));
                if (resid)  v += resid[(size_t)mr * N + nc];
                if (outF) outF[(size_t)mr * N + nc] = v;
                if (outH) outH[(size_t)mr * N + nc] = f2bf(v);
            }
        }
}

// Flash attention: one wave per (b, h, 16 query rows); causal, online softmax.
// Q/V/Y: bf16 [B,T,C] (head offset h*D); Ktr: bf16 [B,H,D,T] (pre-transposed).
#define PLD 40   // 16x32 P tile, padded (80B pitch, 16B aligned)

__global__ __launch_bounds__(32) void k_attn(
    const unsigned short* __restrict__ Q, const unsigned short* __restrict__ Ktr,
    const unsigned short* __restrict__ Vx, unsigned short* __restrict__ Y) {
    __shared__ unsigned short pbuf[16 * PLD];
    int q0   = blockIdx.x * 16;
    int h    = blockIdx.y;
    int b    = blockIdx.z;
    int lane = threadIdx.x;
    int half = lane >> 4;
    int nl   = lane & 15;
    size_t base  = ((size_t)b * TGPT) * CGPT + (size_t)h * DGPT;       // + t*C + d
    size_t kbase = (((size_t)b * HGPT + h) * DGPT) * TGPT;             // + d*T + t

    v16bf qf[2];
#pragma unroll
    for (int kd = 0; kd < 2; ++kd)
        qf[kd] = load_a_global(Q + base + (size_t)q0 * CGPT + kd * 32, CGPT);

    v8f o[4];
#pragma unroll
    for (int nd = 0; nd < 4; ++nd)
#pragma unroll
        for (int i = 0; i < 8; ++i) o[nd][i] = 0.f;
    float mstat[8], lstat[8];
#pragma unroll
    for (int r = 0; r < 8; ++r) { mstat[r] = -1e30f; lstat[r] = 0.f; }

    const float scl = 0.125f;  // 1/sqrt(64)
    int smax = q0 + 16;        // causal bound on key blocks
    for (int s0 = 0; s0 < smax; s0 += 32) {
        // S = Q @ K^T : B-frag from Ktr (lane = token, elems = d)
        v8f sf[2];
#pragma unroll
        for (int st = 0; st < 2; ++st) {
            v8f c = {};
#pragma unroll
            for (int kd = 0; kd < 2; ++kd) {
                v16bf bk;
                const bf16_t* kp = reinterpret_cast<const bf16_t*>(Ktr) + kbase
                                   + (size_t)(kd * 32 + ((lane >> 4) << 3)) * TGPT + s0 + st * 16 + (lane & 15);
#pragma unroll
                for (int e = 0; e < 8; ++e) { bk[e] = kp[e * TGPT]; bk[8 + e] = kp[(16 + e) * TGPT]; }
                c = wmma_bf16(qf[kd], bk, c);
            }
            sf[st] = c;
        }
        // scale + causal mask
#pragma unroll
        for (int st = 0; st < 2; ++st)
#pragma unroll
            for (int r = 0; r < 8; ++r) {
                int mrow = q0 + r + half * 8;
                int ncol = s0 + st * 16 + nl;
                float v  = sf[st][r] * scl;
                sf[st][r] = (ncol <= mrow) ? v : -1e30f;
            }
        // online softmax per row (row lives across one 16-lane half)
        float alpha[8];
#pragma unroll
        for (int r = 0; r < 8; ++r) {
            float mb = fmaxf(sf[0][r], sf[1][r]);
#pragma unroll
            for (int om = 8; om >= 1; om >>= 1) mb = fmaxf(mb, __shfl_xor(mb, om, 32));
            float mnew = fmaxf(mstat[r], mb);
            float a  = __expf(mstat[r] - mnew);
            float p0 = __expf(sf[0][r] - mnew);
            float p1 = __expf(sf[1][r] - mnew);
            sf[0][r] = p0; sf[1][r] = p1;
            float ls = p0 + p1;
#pragma unroll
            for (int om = 8; om >= 1; om >>= 1) ls += __shfl_xor(ls, om, 32);
            lstat[r] = a * lstat[r] + ls;
            mstat[r] = mnew;
            alpha[r] = a;
        }
#pragma unroll
        for (int nd = 0; nd < 4; ++nd)
#pragma unroll
            for (int r = 0; r < 8; ++r) o[nd][r] *= alpha[r];
        // P (C-layout) -> LDS -> A-layout
#pragma unroll
        for (int st = 0; st < 2; ++st)
#pragma unroll
            for (int r = 0; r < 8; ++r)
                pbuf[(r + half * 8) * PLD + st * 16 + nl] = f2bf(sf[st][r]);
        __syncthreads();
        v16bf pf;
        {
            const bf16_t* pp = reinterpret_cast<const bf16_t*>(pbuf)
                               + (lane & 15) * PLD + ((lane >> 4) << 3);
#pragma unroll
            for (int e = 0; e < 8; ++e) { pf[e] = pp[e]; pf[8 + e] = pp[16 + e]; }
        }
        // O += P @ V ; V B-frag: lane = d, elems = tokens
#pragma unroll
        for (int nd = 0; nd < 4; ++nd) {
            v16bf vf;
            const bf16_t* vp = reinterpret_cast<const bf16_t*>(
                Vx + base + (size_t)(s0 + ((lane >> 4) << 3)) * CGPT + nd * 16 + (lane & 15));
#pragma unroll
            for (int e = 0; e < 8; ++e) { vf[e] = vp[e * CGPT]; vf[8 + e] = vp[(16 + e) * CGPT]; }
            o[nd] = wmma_bf16(pf, vf, o[nd]);
        }
        __syncthreads();
    }
    // normalize + store bf16 into [B,T,C]
#pragma unroll
    for (int nd = 0; nd < 4; ++nd)
#pragma unroll
        for (int r = 0; r < 8; ++r) {
            int mrow = q0 + r + half * 8;
            float v  = o[nd][r] / lstat[r];
            Y[base + (size_t)mrow * CGPT + nd * 16 + nl] = f2bf(v);
        }
}

// ---------- host ----------

extern "C" void kernel_launch(void* const* d_in, const int* in_sizes, int n_in,
                              void* d_out, int out_size, void* d_ws, size_t ws_size,
                              hipStream_t stream) {
    (void)in_sizes; (void)n_in; (void)out_size; (void)ws_size;
    const int*   idx  = (const int*)  d_in[0];
    const float* tok  = (const float*)d_in[1];
    const float* pos  = (const float*)d_in[2];
    const float* ln1w = (const float*)d_in[3];
    const float* ln1b = (const float*)d_in[4];
    const float* wq   = (const float*)d_in[5];
    const float* bq   = (const float*)d_in[6];
    const float* wk   = (const float*)d_in[7];
    const float* bk   = (const float*)d_in[8];
    const float* wv   = (const float*)d_in[9];
    const float* bv   = (const float*)d_in[10];
    const float* wo   = (const float*)d_in[11];
    const float* bo   = (const float*)d_in[12];
    const float* ln2w = (const float*)d_in[13];
    const float* ln2b = (const float*)d_in[14];
    const float* wfc  = (const float*)d_in[15];
    const float* bfc  = (const float*)d_in[16];
    const float* wpr  = (const float*)d_in[17];
    const float* bpr  = (const float*)d_in[18];
    const float* lnfw = (const float*)d_in[19];
    const float* lnfb = (const float*)d_in[20];
    const float* whd  = (const float*)d_in[21];

    char* ws = (char*)d_ws;
    size_t off = 0;
    auto alloc = [&](size_t bytes) -> void* {
        void* p = ws + off;
        off += (bytes + 255) & ~(size_t)255;
        return p;
    };
    const size_t wAttnE = (size_t)LGPT * CGPT * CGPT;
    const size_t wMlpE  = (size_t)LGPT * FFGPT * CGPT;
    unsigned short* wq16  = (unsigned short*)alloc(wAttnE * 2);
    unsigned short* wk16  = (unsigned short*)alloc(wAttnE * 2);
    unsigned short* wv16  = (unsigned short*)alloc(wAttnE * 2);
    unsigned short* wo16  = (unsigned short*)alloc(wAttnE * 2);
    unsigned short* wfc16 = (unsigned short*)alloc(wMlpE * 2);
    unsigned short* wpr16 = (unsigned short*)alloc(wMlpE * 2);
    unsigned short* whd16 = (unsigned short*)alloc((size_t)VGPT * CGPT * 2);
    float*          xbuf  = (float*)alloc((size_t)MGPT * CGPT * 4);
    unsigned short* hbuf  = (unsigned short*)alloc((size_t)MGPT * CGPT * 2);
    unsigned short* qbuf  = (unsigned short*)alloc((size_t)MGPT * CGPT * 2);
    unsigned short* kbuf  = (unsigned short*)alloc((size_t)MGPT * CGPT * 2);
    unsigned short* vbuf  = (unsigned short*)alloc((size_t)MGPT * CGPT * 2);
    unsigned short* ktr   = (unsigned short*)alloc((size_t)MGPT * CGPT * 2);
    unsigned short* ybuf  = (unsigned short*)alloc((size_t)MGPT * CGPT * 2);
    unsigned short* mbuf  = (unsigned short*)alloc((size_t)MGPT * FFGPT * 2);

    // weights: fp32 [N,K] -> bf16 transposed [K,N] (per layer in grid.z)
    k_convT<<<dim3(CGPT / 64, CGPT / 64, LGPT), 256, 0, stream>>>(wq, wq16, CGPT, CGPT);
    k_convT<<<dim3(CGPT / 64, CGPT / 64, LGPT), 256, 0, stream>>>(wk, wk16, CGPT, CGPT);
    k_convT<<<dim3(CGPT / 64, CGPT / 64, LGPT), 256, 0, stream>>>(wv, wv16, CGPT, CGPT);
    k_convT<<<dim3(CGPT / 64, CGPT / 64, LGPT), 256, 0, stream>>>(wo, wo16, CGPT, CGPT);
    k_convT<<<dim3(CGPT / 64, FFGPT / 64, LGPT), 256, 0, stream>>>(wfc, wfc16, FFGPT, CGPT);
    k_convT<<<dim3(FFGPT / 64, CGPT / 64, LGPT), 256, 0, stream>>>(wpr, wpr16, CGPT, FFGPT);
    k_convT<<<dim3(CGPT / 64, VGPT / 64, 1),     256, 0, stream>>>(whd, whd16, VGPT, CGPT);

    long nemb = (long)MGPT * CGPT;
    k_embed<<<(int)((nemb + 255) / 256), 256, 0, stream>>>(idx, tok, pos, xbuf, nemb);

    dim3 gC(MGPT / GTM, CGPT / GTN);     // N = 512
    dim3 gF(MGPT / GTM, FFGPT / GTN);    // N = 2048
    dim3 gA(TGPT / 16, HGPT, BGPT);      // attention
    dim3 gT(TGPT / 64, HGPT, BGPT);      // K transpose

    for (int l = 0; l < LGPT; ++l) {
        const unsigned short* wql = wq16 + (size_t)l * CGPT * CGPT;
        const unsigned short* wkl = wk16 + (size_t)l * CGPT * CGPT;
        const unsigned short* wvl = wv16 + (size_t)l * CGPT * CGPT;
        const unsigned short* wol = wo16 + (size_t)l * CGPT * CGPT;
        const unsigned short* wfl = wfc16 + (size_t)l * FFGPT * CGPT;
        const unsigned short* wpl = wpr16 + (size_t)l * CGPT * FFGPT;

        k_layernorm<<<MGPT, 32, 0, stream>>>(xbuf, ln1w + l * CGPT, ln1b + l * CGPT, hbuf);
        k_gemm<<<gC, 256, 0, stream>>>(hbuf, wql, bq + l * CGPT, nullptr,
                                       nullptr, qbuf, MGPT, CGPT, CGPT, 0);
        k_gemm<<<gC, 256, 0, stream>>>(hbuf, wkl, bk + l * CGPT, nullptr,
                                       nullptr, kbuf, MGPT, CGPT, CGPT, 0);
        k_gemm<<<gC, 256, 0, stream>>>(hbuf, wvl, bv + l * CGPT, nullptr,
                                       nullptr, vbuf, MGPT, CGPT, CGPT, 0);
        k_transpose_k<<<gT, 256, 0, stream>>>(kbuf, ktr);
        k_attn<<<gA, 32, 0, stream>>>(qbuf, ktr, vbuf, ybuf);
        k_gemm<<<gC, 256, 0, stream>>>(ybuf, wol, bo + l * CGPT, xbuf,
                                       xbuf, nullptr, MGPT, CGPT, CGPT, 0);
        k_layernorm<<<MGPT, 32, 0, stream>>>(xbuf, ln2w + l * CGPT, ln2b + l * CGPT, hbuf);
        k_gemm<<<gF, 256, 0, stream>>>(hbuf, wfl, bfc + l * FFGPT, nullptr,
                                       nullptr, mbuf, MGPT, FFGPT, CGPT, 1);
        k_gemm<<<gC, 256, 0, stream>>>(mbuf, wpl, bpr + l * CGPT, xbuf,
                                       xbuf, nullptr, MGPT, CGPT, FFGPT, 0);
    }

    k_layernorm<<<MGPT, 32, 0, stream>>>(xbuf, lnfw, lnfb, hbuf);
    dim3 gH(MGPT / GTM, VGPT / GTN);
    k_gemm<<<gH, 256, 0, stream>>>(hbuf, whd16, nullptr, nullptr,
                                   (float*)d_out, nullptr, MGPT, VGPT, CGPT, 0);
}
